// SudokuSolver_15839839388369
// MI455X (gfx1250) — matrix operand order repaired and checked by reading, not compiled, not run on hardware
//
#include <hip/hip_runtime.h>
#include <hip/hip_bf16.h>

typedef __attribute__((ext_vector_type(16))) _Float16 v16h;
typedef __attribute__((ext_vector_type(8)))  _Float16 v8h;
typedef __attribute__((ext_vector_type(8)))  float    v8f;

#define NDIG 9
#define NCELL 81
#define HID  128
#define MT   96     // 81 rows padded to 6 WMMA tiles
#define K1   32     // 27 padded
#define HSTR 136    // h row stride in f16 (272B, 16B aligned, bank-skewed)
#define YSTR 17     // y row stride in f32 (bank-skewed)

__global__ __launch_bounds__(64)
void SudokuSolver_15839839388369_kernel(const float* __restrict__ x,
                                        const float* __restrict__ W1,
                                        const float* __restrict__ W2,
                                        float* __restrict__ out)
{
    __shared__ _Float16 W1s[HID][K1];        // B^T for GEMM1: [N=128][K=32]
    __shared__ _Float16 W2s[16][HID];        // B^T for GEMM2: [N=16pad][K=128]
    __shared__ _Float16 fS[MT][K1];          // feature matrix (A of GEMM1)
    __shared__ _Float16 hS[MT][HSTR];        // hidden acts (A of GEMM2)
    __shared__ float    yS[MT][YSTR];        // logits
    __shared__ float    xp[NCELL * NDIG];    // x_pred (output state)
    __shared__ _Float16 sumsF[256];          // [3][9][9] counts; 243..255 = 0 sentinel
    __shared__ unsigned short fIdx[MT * K1]; // gather table: fS flat -> sumsF index
    __shared__ unsigned emptyF[NCELL];
    __shared__ float    bscore[64];
    __shared__ unsigned bpack[64];

    const int tid  = threadIdx.x;
    const int lane = tid & 31;
    const int wv   = tid >> 5;           // wave 0/1: owns M-tiles {3w..3w+2}
    const int half = lane >> 4;          // lane-half for WMMA fragment layout
    const int l16  = lane & 15;
    const int b    = blockIdx.x;
    const float* xb = x + (size_t)b * (NCELL * NDIG);

    // ------------- one-time init: weights -> f16 LDS, board state -------------
    for (int i = tid; i < HID * K1; i += 64) {
        int n = i >> 5, k = i & 31;
        W1s[n][k] = (k < 27) ? (_Float16)W1[n * 27 + k] : (_Float16)0.f;
    }
    for (int i = tid; i < 16 * HID; i += 64) {
        int n = i >> 7, k = i & 127;
        W2s[n][k] = (n < 9) ? (_Float16)W2[n * HID + k] : (_Float16)0.f;
    }
    for (int i = tid; i < NCELL * NDIG; i += 64) xp[i] = xb[i];
    for (int q = tid; q < NCELL; q += 64) {
        float s = 0.f;
        #pragma unroll
        for (int e = 0; e < 9; ++e) s += xb[q * 9 + e];
        emptyF[q] = (s == 0.f) ? 1u : 0u;
    }
    for (int s = tid; s < 256; s += 64) {        // 3 types x 9 groups x 9 digits
        float acc = 0.f;
        if (s < 243) {
            int t = s / 81, g = (s / 9) % 9, e = s % 9;
            #pragma unroll
            for (int i = 0; i < 9; ++i) {
                int q;
                if (t == 0)      q = g * 9 + i;                                   // row
                else if (t == 1) q = i * 9 + g;                                   // col
                else             q = ((g / 3) * 3 + i / 3) * 9 + (g % 3) * 3 + i % 3; // box
                acc += xb[q * 9 + e];
            }
        }
        sumsF[s] = (_Float16)acc;
    }
    // gather table for the reference's row-major reshape of cons[3,81,9] -> f[81,27]
    for (int i = tid; i < MT * K1; i += 64) {
        int Q = i >> 5, j = i & 31;
        unsigned short idx = 243;           // sentinel -> 0.0 (padding)
        if (Q < NCELL && j < 27) {
            int r = Q / 27;                 // constraint type
            int m = Q % 27;
            int q = 3 * m + j / 9;          // actual cell
            int e = j % 9;
            int g;
            if (r == 0)      g = q / 9;
            else if (r == 1) g = q % 9;
            else             g = (q / 27) * 3 + (q % 9) / 3;
            idx = (unsigned short)(r * 81 + g * 9 + e);
        }
        fIdx[i] = idx;
    }
    __syncthreads();

    // W2 B-fragments are invariant over all 81 steps: pin them in registers
    v16h w2f[4];
    #pragma unroll
    for (int kc = 0; kc < 4; ++kc) {
        v8h blo = *(const v8h*)&W2s[l16][kc * 32 + half * 8];
        v8h bhi = *(const v8h*)&W2s[l16][kc * 32 + 16 + half * 8];
        #pragma unroll
        for (int i = 0; i < 8; ++i) { w2f[kc][i] = blo[i]; w2f[kc][i + 8] = bhi[i]; }
    }

    // ----------------------------- 81 solver steps ----------------------------
    for (int step = 0; step < NCELL; ++step) {
        // Phase A: materialize f via pure LDS gather (no per-step index math)
        for (int i = tid; i < MT * K1; i += 64)
            ((_Float16*)fS)[i] = sumsF[fIdx[i]];
        __syncthreads();

        // Phase B: GEMM1  h = relu(f @ W1^T)   (nt-outer: each B frag loaded once)
        v16h afrag[3];
        #pragma unroll
        for (int mi = 0; mi < 3; ++mi) {
            const int arow = (wv * 3 + mi) * 16 + l16;
            v8h alo = *(const v8h*)&fS[arow][half * 8];
            v8h ahi = *(const v8h*)&fS[arow][16 + half * 8];
            #pragma unroll
            for (int i = 0; i < 8; ++i) { afrag[mi][i] = alo[i]; afrag[mi][i + 8] = ahi[i]; }
        }
        for (int nt = 0; nt < 8; ++nt) {
            const int bcol = nt * 16 + l16;
            v8h blo = *(const v8h*)&W1s[bcol][half * 8];
            v8h bhi = *(const v8h*)&W1s[bcol][16 + half * 8];
            v16h bb;
            #pragma unroll
            for (int i = 0; i < 8; ++i) { bb[i] = blo[i]; bb[i + 8] = bhi[i]; }
            #pragma unroll
            for (int mi = 0; mi < 3; ++mi) {
                v8f c = {};
                c = __builtin_amdgcn_wmma_f32_16x16x32_f16(false, afrag[mi], false, bb,
                                                           (short)0, c, false, false);
                const int mbase = (wv * 3 + mi) * 16 + half * 8;
                #pragma unroll
                for (int i = 0; i < 8; ++i) {
                    float v = c[i] > 0.f ? c[i] : 0.f;          // relu
                    hS[mbase + i][bcol] = (_Float16)v;
                }
            }
        }
        // no barrier: each wave consumes only the h-rows it produced

        // Phase C: GEMM2  y = h @ W2^T  (K=128 in 4 chained WMMAs, pinned B frags)
        #pragma unroll
        for (int mi = 0; mi < 3; ++mi) {
            const int arow = (wv * 3 + mi) * 16 + l16;
            v8f c = {};
            #pragma unroll
            for (int kc = 0; kc < 4; ++kc) {
                v8h alo = *(const v8h*)&hS[arow][kc * 32 + half * 8];
                v8h ahi = *(const v8h*)&hS[arow][kc * 32 + 16 + half * 8];
                v16h a;
                #pragma unroll
                for (int i = 0; i < 8; ++i) { a[i] = alo[i]; a[i + 8] = ahi[i]; }
                c = __builtin_amdgcn_wmma_f32_16x16x32_f16(false, a, false, w2f[kc],
                                                           (short)0, c, false, false);
            }
            const int mbase = (wv * 3 + mi) * 16 + half * 8;
            #pragma unroll
            for (int i = 0; i < 8; ++i) yS[mbase + i][l16] = c[i];
        }
        __syncthreads();

        // Phase D: per-cell softmax over digits + x_pred update + local argmax
        float    myScore = 0.f;
        unsigned myPack  = 0u;
        for (int q = tid; q < NCELL; q += 64) {
            if (emptyF[q]) {
                float v[9], m = -1e30f;
                #pragma unroll
                for (int e = 0; e < 9; ++e) { v[e] = yS[q][e]; m = fmaxf(m, v[e]); }
                float s = 0.f, mx = 0.f; int pos = 0;
                #pragma unroll
                for (int e = 0; e < 9; ++e) {
                    float ex = __expf(v[e] - m);
                    v[e] = ex; s += ex;
                    if (ex > mx) { mx = ex; pos = e; }
                }
                float rs = __builtin_amdgcn_rcpf(s);
                #pragma unroll
                for (int e = 0; e < 9; ++e) xp[q * 9 + e] = v[e] * rs;
                float sc = mx * rs;
                if (sc > myScore) { myScore = sc; myPack = (unsigned)(q * 16 + pos); }
            }
        }
        bscore[tid] = myScore;
        bpack[tid]  = myPack;
        __syncthreads();

        // Phase E: commit most-confident digit, bump constraint counters
        if (tid == 0) {
            float best = 0.f; unsigned bp = 0u;
            for (int i = 0; i < 64; ++i)
                if (bscore[i] > best) { best = bscore[i]; bp = bpack[i]; }
            if (best > 0.f) {              // best==0 <=> no empty cells left
                int q = (int)(bp >> 4), d = (int)(bp & 15u);
                emptyF[q] = 0u;
                int row = q / 9, col = q % 9, box = (row / 3) * 3 + col / 3;
                sumsF[0 * 81 + row * 9 + d] = sumsF[0 * 81 + row * 9 + d] + (_Float16)1.f;
                sumsF[1 * 81 + col * 9 + d] = sumsF[1 * 81 + col * 9 + d] + (_Float16)1.f;
                sumsF[2 * 81 + box * 9 + d] = sumsF[2 * 81 + box * 9 + d] + (_Float16)1.f;
            }
        }
        __syncthreads();
    }

    // ------------------------------- write back -------------------------------
    float* ob = out + (size_t)b * (NCELL * NDIG);
    for (int i = tid; i < NCELL * NDIG; i += 64) ob[i] = xp[i];
}

extern "C" void kernel_launch(void* const* d_in, const int* in_sizes, int n_in,
                              void* d_out, int out_size, void* d_ws, size_t ws_size,
                              hipStream_t stream) {
    (void)n_in; (void)out_size; (void)d_ws; (void)ws_size;
    const float* x  = (const float*)d_in[0];
    const float* W1 = (const float*)d_in[1];
    const float* W2 = (const float*)d_in[2];
    // d_in[3] = constraint_mask: structure is hardcoded (row/col/box counters)
    const int B = in_sizes[0] / (NCELL * NDIG);
    SudokuSolver_15839839388369_kernel<<<B, 64, 0, stream>>>(x, W1, W2, (float*)d_out);
}
